// PairformerLite_77807627534830
// MI455X (gfx1250) — matrix operand order repaired and checked by reading, not compiled, not run on hardware
//
#include <hip/hip_runtime.h>
#include <hip/hip_bf16.h>

// ---------------------------------------------------------------------------
// Pairformer-lite for gfx1250 (MI455X): bf16 WMMA (v_wmma_f32_16x16x32_bf16)
// for all GEMMs; fp32 accumulate; double-buffered async global->LDS staging
// (global_load_async_to_lds_b128 + s_wait_asynccnt) in the einsum kernel.
// ---------------------------------------------------------------------------

typedef __bf16 bf16;
typedef __attribute__((ext_vector_type(16))) __bf16 v16bf;
typedef __attribute__((ext_vector_type(8)))  __bf16 v8bf;
typedef __attribute__((ext_vector_type(8)))  float  v8f;
typedef __attribute__((ext_vector_type(4)))  float  f32x4;

static constexpr int   NSEQ = 512;
static constexpr int   SDIM = 64;
static constexpr int   CDIM = 32;
static constexpr long  NN   = (long)NSEQ * NSEQ;           // 262144
static constexpr float EPSV = 1e-5f;

__device__ __forceinline__ v8f vzero8() {
    v8f z = {0.f, 0.f, 0.f, 0.f, 0.f, 0.f, 0.f, 0.f};
    return z;
}

// Build a 16x32-bf16 A fragment lane-slice from two contiguous 8-half chunks.
__device__ __forceinline__ v16bf frag_cat(const bf16* p0, const bf16* p1) {
    v8bf lo = *reinterpret_cast<const v8bf*>(p0);
    v8bf hi = *reinterpret_cast<const v8bf*>(p1);
    v16bf r;
#pragma unroll
    for (int e = 0; e < 8; ++e) { r[e] = lo[e]; r[8 + e] = hi[e]; }
    return r;
}

__device__ __forceinline__ v8f wmma_bf16(v16bf a, v16bf b, v8f c) {
    // 8 args: (neg_a, A, neg_b, B, c_mod, C, reuse_a, reuse_b)
    return __builtin_amdgcn_wmma_f32_16x16x32_bf16(false, a, false, b,
                                                   (short)0, c, false, false);
}

__device__ __forceinline__ float sigmoidf_(float v) {
    return 1.f / (1.f + __expf(-v));
}

// CDNA5 async global->LDS copy (16B per lane), tracked by ASYNCcnt.
// VDST = per-lane LDS byte offset, VADDR = per-lane 64-bit global address.
__device__ __forceinline__ void async_ld_b128(unsigned lds_off, const void* gaddr) {
    asm volatile("global_load_async_to_lds_b128 %0, %1, off"
                 :: "v"(lds_off), "v"(gaddr)
                 : "memory");
}

__device__ __forceinline__ void wait_asynccnt0() {
    asm volatile("s_wait_asynccnt 0x0" ::: "memory");
}

// ---------------------------------------------------------------------------
// K0: recycle add.  sbuf = single + ps*prev_single ; pbuf = pair + pp*prev_pair
// ---------------------------------------------------------------------------
__global__ __launch_bounds__(256) void k_recycle(
    const float* __restrict__ single, const float* __restrict__ pair,
    const float* __restrict__ psin,   const float* __restrict__ ppair,
    const float* __restrict__ ps_s,   const float* __restrict__ pp_s,
    float* __restrict__ sbuf, float* __restrict__ pbuf)
{
    long i = (long)blockIdx.x * 256 + threadIdx.x;
    const long NS = (long)NSEQ * SDIM;
    if (i < NS) {
        sbuf[i] = single[i] + ps_s[0] * psin[i];
    } else {
        long j = i - NS;                     // j < NN*C by grid construction
        pbuf[j] = pair[j] + pp_s[0] * ppair[j];
    }
}

// ---------------------------------------------------------------------------
// K1: triangle LN + 6 projections.  Wave = one pair position, lane = channel.
// Stores bf16 operand planes (channel-major, pre-transposed for the einsum)
// and fp32 sigmoid gates.
// ---------------------------------------------------------------------------
__global__ __launch_bounds__(256) void k_tri_lnproj(
    const float* __restrict__ pbuf,
    const float* __restrict__ ng,  const float* __restrict__ nb,
    const float* __restrict__ wao, const float* __restrict__ wbo,
    const float* __restrict__ wgo, const float* __restrict__ bgo,
    const float* __restrict__ wai, const float* __restrict__ wbi,
    const float* __restrict__ wgi, const float* __restrict__ bgi,
    bf16* __restrict__ aoP, bf16* __restrict__ boT,
    bf16* __restrict__ aiP, bf16* __restrict__ biT,
    float* __restrict__ go, float* __restrict__ gi)
{
    __shared__ float xsh[8][32];
    const int wave = threadIdx.x >> 5, lane = threadIdx.x & 31;
    const long pos = (long)blockIdx.x * 8 + wave;
    const int p = (int)(pos >> 9), q = (int)(pos & 511);

    float x = pbuf[pos * 32 + lane];
    float s = x;
#pragma unroll
    for (int o = 16; o; o >>= 1) s += __shfl_xor(s, o);
    float m = s * (1.f / 32.f);
    float d = x - m;
    float v = d * d;
#pragma unroll
    for (int o = 16; o; o >>= 1) v += __shfl_xor(v, o);
    float xn = d * rsqrtf(v * (1.f / 32.f) + EPSV) * ng[lane] + nb[lane];
    xsh[wave][lane] = xn;
    __syncthreads();

    float ao = 0.f, bo = 0.f, g_o = 0.f, ai = 0.f, bi = 0.f, g_i = 0.f;
#pragma unroll 8
    for (int c = 0; c < 32; ++c) {
        float xc = xsh[wave][c];           // LDS broadcast
        int wi = c * 32 + lane;
        ao += xc * wao[wi]; bo += xc * wbo[wi]; g_o += xc * wgo[wi];
        ai += xc * wai[wi]; bi += xc * wbi[wi]; g_i += xc * wgi[wi];
    }

    const long pq = (long)p * 512 + q, qp = (long)q * 512 + p;
    aoP[(long)lane * NN + pq] = (bf16)ao;   // A_c[i=p][k=q]
    boT[(long)lane * NN + qp] = (bf16)bo;   // (B_c)^T[j=q][k=p]
    aiP[(long)lane * NN + qp] = (bf16)ai;   // P_c[i=q][k=p]
    biT[(long)lane * NN + pq] = (bf16)bi;   // (Q_c)^T[j=p][k=q]
    go[pos * 32 + lane] = sigmoidf_(g_o + bgo[lane]);
    gi[pos * 32 + lane] = sigmoidf_(g_i + bgi[lane]);
}

// ---------------------------------------------------------------------------
// K2: per-channel 512x512x512 bf16 GEMMs (both einsum directions), gates
// applied, scaled by 1/bsz.  128x64 tile per 256-thread block (8 waves,
// each wave a 32x32 quadrant via 2x2 WMMA frags).  Double-buffered LDS fed
// by async global->LDS copies; one barrier + s_wait_asynccnt per K step.
// Two statically-indexed phase loops (no dynamic acc indexing -> no spills).
// Grid: (32 tiles, 32 channels).
// ---------------------------------------------------------------------------
__global__ __launch_bounds__(256) void k_tri_einsum(
    const bf16* __restrict__ aoP, const bf16* __restrict__ boT,
    const bf16* __restrict__ aiP, const bf16* __restrict__ biT,
    const float* __restrict__ go, const float* __restrict__ gi,
    float* __restrict__ gt)
{
    __shared__ __align__(16) bf16 lA[2][128][32];   // 16 KB
    __shared__ __align__(16) bf16 lB[2][64][32];    //  8 KB

    const int c  = blockIdx.y;
    const int J0 = (blockIdx.x & 7) * 64;
    const int I0 = (blockIdx.x >> 3) * 128;
    const long plane = (long)c * NN;

    const int tid  = threadIdx.x;
    const int wave = tid >> 5, lane = tid & 31;
    const int l15  = lane & 15;
    const int hiA  = (lane >> 4) * 16;    // A K-base per half-wave
    const int hiB  = (lane >> 4) * 16;    // B K-base per half-wave
    const int mrow = (wave & 3) * 32;
    const int ncol = (wave >> 2) * 32;

    const bf16* Apl[2] = {aoP + plane, aiP + plane};
    const bf16* Bpl[2] = {boT + plane, biT + plane};

    // Per-thread async copy geometry: A = 512 16B chunks (2/thread),
    // B = 256 16B chunks (1/thread).  Global step: [0,32) = 2 phases x 16.
    auto issue = [&](int step, int buf) {
        const int ph = step >> 4;
        const int k0 = (step & 15) * 32;
        const bf16* Ap = Apl[ph];
        const bf16* Bp = Bpl[ph];
#pragma unroll
        for (int ld = 0; ld < 2; ++ld) {
            int ch = tid + ld * 256;
            int row = ch >> 2, kc = (ch & 3) * 8;
            async_ld_b128((unsigned)(uintptr_t)&lA[buf][row][kc],
                          (const void*)(Ap + (long)(I0 + row) * 512 + k0 + kc));
        }
        {
            int row = tid >> 2, kc = (tid & 3) * 8;
            async_ld_b128((unsigned)(uintptr_t)&lB[buf][row][kc],
                          (const void*)(Bp + (long)(J0 + row) * 512 + k0 + kc));
        }
    };

    auto prefetch2 = [&](int step) {       // L2 prefetch two tiles ahead
        if (step < 32) {
            const int ph2 = step >> 4;
            const int k2  = (step & 15) * 32;
            __builtin_prefetch((const void*)(Apl[ph2] + (long)(I0 + (tid >> 1)) * 512 + k2), 0, 1);
            __builtin_prefetch((const void*)(Bpl[ph2] + (long)(J0 + (tid >> 2)) * 512 + k2), 0, 1);
        }
    };

    auto step_body = [&](int step, v8f (&acc)[2][2]) {
        const int buf = step & 1;
        wait_asynccnt0();       // this wave's copies into `buf` have landed
        __syncthreads();        // everyone's copies landed; prev compute done
        if (step + 1 < 32) issue(step + 1, buf ^ 1);
        prefetch2(step + 2);

        v16bf Af[2], Bf[2];
#pragma unroll
        for (int mm = 0; mm < 2; ++mm) {
            const bf16* r = &lA[buf][mrow + mm * 16 + l15][hiA];
            Af[mm] = frag_cat(r, r + 8);
        }
#pragma unroll
        for (int nn = 0; nn < 2; ++nn)
            Bf[nn] = *reinterpret_cast<const v16bf*>(&lB[buf][ncol + nn * 16 + l15][hiB]);

#pragma unroll
        for (int mm = 0; mm < 2; ++mm)
#pragma unroll
            for (int nn = 0; nn < 2; ++nn)
                acc[mm][nn] = wmma_bf16(Af[mm], Bf[nn], acc[mm][nn]);
    };

    // Statically-indexed accumulators: keeps all 8 C-fragments in VGPRs.
    v8f acc_o[2][2], acc_i[2][2];
#pragma unroll
    for (int mm = 0; mm < 2; ++mm)
#pragma unroll
        for (int nn = 0; nn < 2; ++nn) { acc_o[mm][nn] = vzero8(); acc_i[mm][nn] = vzero8(); }

    issue(0, 0);
#pragma unroll 1
    for (int step = 0; step < 16; ++step)       // phase 0: t_o
        step_body(step, acc_o);
#pragma unroll 1
    for (int step = 16; step < 32; ++step)      // phase 1: t_i
        step_body(step, acc_i);

    const float inv = 1.f / 512.f;          // bsz = pair.shape[0] = 512
    const int rofs = (lane >= 16) ? 8 : 0;
#pragma unroll
    for (int mm = 0; mm < 2; ++mm)
#pragma unroll
        for (int nn = 0; nn < 2; ++nn) {
            int col = J0 + ncol + nn * 16 + l15;
#pragma unroll
            for (int r = 0; r < 8; ++r) {
                int row = I0 + mrow + mm * 16 + r + rofs;
                long idx = ((long)row * 512 + col) * 32 + c;
                gt[idx] = (go[idx] * acc_o[mm][nn][r] +
                           gi[idx] * acc_i[mm][nn][r]) * inv;
            }
        }
}

// ---------------------------------------------------------------------------
// K3: pair += gt @ wo + bo   (WMMA, K=32).  Wave = 16 positions.
// ---------------------------------------------------------------------------
__global__ __launch_bounds__(256) void k_tri_out(
    const float* __restrict__ gt, const float* __restrict__ wo,
    const float* __restrict__ bo, float* __restrict__ pair)
{
    __shared__ __align__(16) bf16 xA[8][16][32];
    const int wave = threadIdx.x >> 5, lane = threadIdx.x & 31;
    const int l15 = lane & 15, h = lane >> 4;
    const long pos0 = (long)blockIdx.x * 128 + wave * 16;

    const float* src = gt + (pos0 + l15) * 32 + h * 16;
#pragma unroll
    for (int e = 0; e < 16; ++e) xA[wave][l15][h * 16 + e] = (bf16)src[e];
    __syncthreads();

    v16bf Af = frag_cat(&xA[wave][l15][h * 16], &xA[wave][l15][h * 16 + 8]);

    v8f accn[2] = {vzero8(), vzero8()};
#pragma unroll
    for (int n = 0; n < 2; ++n) {
        v16bf Bf;
        int o = n * 16 + l15, kb = h * 16;
#pragma unroll
        for (int e = 0; e < 16; ++e) Bf[e] = (bf16)wo[(kb + e) * 32 + o];
        accn[n] = wmma_bf16(Af, Bf, accn[n]);
    }
#pragma unroll
    for (int n = 0; n < 2; ++n) {
        int o = n * 16 + l15;
        float bias = bo[o];
#pragma unroll
        for (int r = 0; r < 8; ++r) {
            long pos = pos0 + r + h * 8;
            pair[pos * 32 + o] += accn[n][r] + bias;
        }
    }
}

// ---------------------------------------------------------------------------
// K4: pair += FF(pair)  (LN -> WMMA 32->128 -> SiLU -> WMMA 128->32)
// ---------------------------------------------------------------------------
__global__ __launch_bounds__(256) void k_pair_ff(
    float* __restrict__ pair,
    const float* __restrict__ ng, const float* __restrict__ nb,
    const float* __restrict__ w1, const float* __restrict__ b1,
    const float* __restrict__ w2, const float* __restrict__ b2)
{
    __shared__ __align__(16) bf16 xA[8][16][32];     // 8 KB
    __shared__ __align__(16) bf16 hA[8][16][128];    // 32 KB
    __shared__ __align__(16) bf16 wB1T[128][32];     // [o][k], 8 KB
    __shared__ __align__(16) bf16 wB2T[32][128];     // [o][k], 8 KB

    const int tid = threadIdx.x, wave = tid >> 5, lane = tid & 31;
    const int l15 = lane & 15, h = lane >> 4;

    for (int e = tid; e < 32 * 128; e += 256) {      // w1[k][o] -> wB1T[o][k]
        int k = e >> 7, o = e & 127;
        wB1T[o][k] = (bf16)w1[e];
    }
    for (int e = tid; e < 128 * 32; e += 256) {      // w2[k][o] -> wB2T[o][k]
        int k = e >> 5, o = e & 31;
        wB2T[o][k] = (bf16)w2[e];
    }

    const long pos0 = (long)blockIdx.x * 128 + wave * 16;

    // LayerNorm: lane pair (h=0/1) covers one row of 32 channels.
    float xs[16];
    const float* src = pair + (pos0 + l15) * 32 + h * 16;
    float s = 0.f, s2 = 0.f;
#pragma unroll
    for (int e = 0; e < 16; ++e) { xs[e] = src[e]; s += xs[e]; s2 += xs[e] * xs[e]; }
    s += __shfl_xor(s, 16);
    s2 += __shfl_xor(s2, 16);
    float m = s * (1.f / 32.f);
    float var = s2 * (1.f / 32.f) - m * m;
    float rs = rsqrtf(var + EPSV);
#pragma unroll
    for (int e = 0; e < 16; ++e)
        xA[wave][l15][h * 16 + e] =
            (bf16)((xs[e] - m) * rs * ng[h * 16 + e] + nb[h * 16 + e]);
    __syncthreads();

    v16bf Af = frag_cat(&xA[wave][l15][h * 16], &xA[wave][l15][h * 16 + 8]);

    // GEMM1: 16x128 = 8 WMMA tiles, then SiLU into hA (bf16).
#pragma unroll
    for (int n = 0; n < 8; ++n) {
        int o = n * 16 + l15;
        v16bf Bf = *reinterpret_cast<const v16bf*>(&wB1T[o][h * 16]);
        v8f acc = vzero8();
        acc = wmma_bf16(Af, Bf, acc);
#pragma unroll
        for (int r = 0; r < 8; ++r) {
            int row = r + h * 8;
            float v = acc[r] + b1[o];
            hA[wave][row][o] = (bf16)(v * sigmoidf_(v));
        }
    }
    __syncthreads();

    // GEMM2: K=128 in 4 chunks of 32.
    v8f acc2[2] = {vzero8(), vzero8()};
#pragma unroll
    for (int kk = 0; kk < 4; ++kk) {
        v16bf Af2 = frag_cat(&hA[wave][l15][kk * 32 + h * 16],
                             &hA[wave][l15][kk * 32 + h * 16 + 8]);
#pragma unroll
        for (int n = 0; n < 2; ++n) {
            int o = n * 16 + l15;
            v16bf Bf2 = *reinterpret_cast<const v16bf*>(&wB2T[o][kk * 32 + h * 16]);
            acc2[n] = wmma_bf16(Af2, Bf2, acc2[n]);
        }
    }
#pragma unroll
    for (int n = 0; n < 2; ++n) {
        int o = n * 16 + l15;
        float bias = b2[o];
#pragma unroll
        for (int r = 0; r < 8; ++r) {
            long pos = pos0 + r + h * 8;
            pair[pos * 32 + o] += acc2[n][r] + bias;
        }
    }
}

// ---------------------------------------------------------------------------
// K5: pooled[p][c] = mean_q LN(pair[p,q,:])[c]
// ---------------------------------------------------------------------------
__global__ __launch_bounds__(256) void k_pooled(
    const float* __restrict__ pair,
    const float* __restrict__ ng, const float* __restrict__ nb,
    float* __restrict__ pooled)
{
    __shared__ float red[8][32];
    const int p = blockIdx.x;
    const int wave = threadIdx.x >> 5, lane = threadIdx.x & 31;
    const float g = ng[lane], b = nb[lane];
    float acc = 0.f;
    for (int q = wave; q < 512; q += 8) {
        float x = pair[((long)p * 512 + q) * 32 + lane];
        float s = x;
#pragma unroll
        for (int o = 16; o; o >>= 1) s += __shfl_xor(s, o);
        float m = s * (1.f / 32.f);
        float d = x - m;
        float v = d * d;
#pragma unroll
        for (int o = 16; o; o >>= 1) v += __shfl_xor(v, o);
        acc += d * rsqrtf(v * (1.f / 32.f) + EPSV) * g + b;
    }
    red[wave][lane] = acc;
    __syncthreads();
    if (wave == 0) {
        float t = 0.f;
#pragma unroll
        for (int w = 0; w < 8; ++w) t += red[w][lane];
        pooled[p * 32 + lane] = t * (1.f / 512.f);
    }
}

// ---------------------------------------------------------------------------
// K6: single += s_scale * (pooled @ p2s_w + p2s_b)
// ---------------------------------------------------------------------------
__global__ __launch_bounds__(256) void k_p2s(
    const float* __restrict__ pooled, const float* __restrict__ w,
    const float* __restrict__ b, const float* __restrict__ s_scale,
    float* __restrict__ sbuf)
{
    int idx = blockIdx.x * 256 + threadIdx.x;   // p*64 + s
    int p = idx >> 6, s = idx & 63;
    float acc = b[s];
#pragma unroll 8
    for (int c = 0; c < 32; ++c) acc += pooled[p * 32 + c] * w[c * 64 + s];
    sbuf[idx] += s_scale[0] * acc;
}

// ---------------------------------------------------------------------------
// K7: single += FF(single)   (S=64, hidden=256; one block per row)
// ---------------------------------------------------------------------------
__global__ __launch_bounds__(64) void k_single_ff(
    float* __restrict__ sbuf,
    const float* __restrict__ ng, const float* __restrict__ nb,
    const float* __restrict__ w1, const float* __restrict__ b1,
    const float* __restrict__ w2, const float* __restrict__ b2)
{
    __shared__ float xs[64];
    __shared__ float hs[256];
    const int p = blockIdx.x, t = threadIdx.x;
    float x = sbuf[p * 64 + t];
    xs[t] = x;
    __syncthreads();
    float s = 0.f, s2 = 0.f;
#pragma unroll 8
    for (int k = 0; k < 64; ++k) { float v = xs[k]; s += v; s2 += v * v; }
    float m = s * (1.f / 64.f);
    float var = s2 * (1.f / 64.f) - m * m;
    float xn = (x - m) * rsqrtf(var + EPSV) * ng[t] + nb[t];
    __syncthreads();
    xs[t] = xn;
    __syncthreads();
#pragma unroll
    for (int j = 0; j < 4; ++j) {
        int o = t + 64 * j;
        float a = b1[o];
#pragma unroll 8
        for (int k = 0; k < 64; ++k) a += xs[k] * w1[k * 256 + o];
        hs[o] = a * sigmoidf_(a);
    }
    __syncthreads();
    float a = b2[t];
#pragma unroll 8
    for (int j = 0; j < 256; ++j) a += hs[j] * w2[j * 64 + t];
    sbuf[p * 64 + t] = x + a;
}

// ---------------------------------------------------------------------------
// Host-side launcher
// ---------------------------------------------------------------------------
extern "C" void kernel_launch(void* const* d_in, const int* in_sizes, int n_in,
                              void* d_out, int out_size, void* d_ws, size_t ws_size,
                              hipStream_t stream) {
    const float* in_single = (const float*)d_in[0];
    const float* in_pair   = (const float*)d_in[1];
    const float* in_psin   = (const float*)d_in[2];
    const float* in_ppair  = (const float*)d_in[3];
    const float* tri_ng  = (const float*)d_in[4];
    const float* tri_nb  = (const float*)d_in[5];
    const float* tri_wao = (const float*)d_in[6];
    const float* tri_wbo = (const float*)d_in[7];
    const float* tri_wgo = (const float*)d_in[8];
    const float* tri_bgo = (const float*)d_in[9];
    const float* tri_wai = (const float*)d_in[10];
    const float* tri_wbi = (const float*)d_in[11];
    const float* tri_wgi = (const float*)d_in[12];
    const float* tri_bgi = (const float*)d_in[13];
    const float* tri_wo  = (const float*)d_in[14];
    const float* tri_bo  = (const float*)d_in[15];
    const float* pff_ng  = (const float*)d_in[16];
    const float* pff_nb  = (const float*)d_in[17];
    const float* pff_w1  = (const float*)d_in[18];
    const float* pff_b1  = (const float*)d_in[19];
    const float* pff_w2  = (const float*)d_in[20];
    const float* pff_b2  = (const float*)d_in[21];
    const float* p2s_ng  = (const float*)d_in[22];
    const float* p2s_nb  = (const float*)d_in[23];
    const float* p2s_w   = (const float*)d_in[24];
    const float* p2s_b   = (const float*)d_in[25];
    const float* sff_ng  = (const float*)d_in[26];
    const float* sff_nb  = (const float*)d_in[27];
    const float* sff_w1  = (const float*)d_in[28];
    const float* sff_b1  = (const float*)d_in[29];
    const float* sff_w2  = (const float*)d_in[30];
    const float* sff_b2  = (const float*)d_in[31];
    const float* s_scale  = (const float*)d_in[32];
    const float* ps_scale = (const float*)d_in[33];
    const float* pp_scale = (const float*)d_in[34];

    // Outputs accumulate in-place in d_out: [single | pair]
    float* sbuf = (float*)d_out;
    float* pbuf = sbuf + (size_t)NSEQ * SDIM;

    // Workspace partition
    char* w = (char*)d_ws;
    bf16* aoP = (bf16*)w; w += (size_t)NN * CDIM * sizeof(bf16);
    bf16* boT = (bf16*)w; w += (size_t)NN * CDIM * sizeof(bf16);
    bf16* aiP = (bf16*)w; w += (size_t)NN * CDIM * sizeof(bf16);
    bf16* biT = (bf16*)w; w += (size_t)NN * CDIM * sizeof(bf16);
    float* go = (float*)w; w += (size_t)NN * CDIM * sizeof(float);
    float* gi = (float*)w; w += (size_t)NN * CDIM * sizeof(float);
    float* gt = (float*)w; w += (size_t)NN * CDIM * sizeof(float);
    float* pooled = (float*)w;

    const long NS = (long)NSEQ * SDIM;
    const long NNC = NN * CDIM;

    k_recycle<<<(unsigned)((NS + NNC + 255) / 256), 256, 0, stream>>>(
        in_single, in_pair, in_psin, in_ppair, ps_scale, pp_scale, sbuf, pbuf);

    for (int i = 0; i < 2; ++i) {
        const int CC = CDIM * CDIM;                 // 1024
        k_tri_lnproj<<<(unsigned)(NN / 8), 256, 0, stream>>>(
            pbuf, tri_ng + i * CDIM, tri_nb + i * CDIM,
            tri_wao + i * CC, tri_wbo + i * CC, tri_wgo + i * CC, tri_bgo + i * CDIM,
            tri_wai + i * CC, tri_wbi + i * CC, tri_wgi + i * CC, tri_bgi + i * CDIM,
            aoP, boT, aiP, biT, go, gi);

        k_tri_einsum<<<dim3(32, 32), 256, 0, stream>>>(aoP, boT, aiP, biT, go, gi, gt);

        k_tri_out<<<(unsigned)(NN / 128), 256, 0, stream>>>(
            gt, tri_wo + i * CC, tri_bo + i * CDIM, pbuf);

        k_pair_ff<<<(unsigned)(NN / 128), 256, 0, stream>>>(
            pbuf, pff_ng + i * CDIM, pff_nb + i * CDIM,
            pff_w1 + i * CDIM * 4 * CDIM, pff_b1 + i * 4 * CDIM,
            pff_w2 + i * 4 * CDIM * CDIM, pff_b2 + i * CDIM);

        k_pooled<<<NSEQ, 256, 0, stream>>>(
            pbuf, p2s_ng + i * CDIM, p2s_nb + i * CDIM, pooled);

        k_p2s<<<(unsigned)(NS / 256), 256, 0, stream>>>(
            pooled, p2s_w + i * CDIM * SDIM, p2s_b + i * SDIM, s_scale, sbuf);

        k_single_ff<<<NSEQ, 64, 0, stream>>>(
            sbuf, sff_ng + i * SDIM, sff_nb + i * SDIM,
            sff_w1 + i * SDIM * 4 * SDIM, sff_b1 + i * 4 * SDIM,
            sff_w2 + i * 4 * SDIM * SDIM, sff_b2 + i * SDIM);
    }
}
